// GCNEncoder_4028679324252
// MI455X (gfx1250) — compile-verified, matrix operand order checked by reading, not run on hardware
//
#include <hip/hip_runtime.h>
#include <stdint.h>

// ---- WMMA / vector types (gfx1250, wave32) ----
typedef __attribute__((ext_vector_type(16))) __bf16    v16bf;
typedef __attribute__((ext_vector_type(8)))  float     v8f;
typedef __attribute__((ext_vector_type(4)))  uint32_t  v4u;
typedef __attribute__((ext_vector_type(4)))  float     v4f;

struct U8 { v4u lo, hi; };   // 32 bytes == one bf16 fragment per lane

// RNE f32 -> bf16, packed pair (low K element in [15:0])
static __device__ __forceinline__ uint32_t pack2(float lo, float hi) {
    uint32_t ul = __builtin_bit_cast(uint32_t, lo);
    uint32_t uh = __builtin_bit_cast(uint32_t, hi);
    ul = (ul + 0x7FFFu + ((ul >> 16) & 1u)) >> 16;
    uh = (uh + 0x7FFFu + ((uh >> 16) & 1u)) >> 16;
    return (ul & 0xFFFFu) | (uh << 16);
}

static __device__ __forceinline__ v4f relu4(v4f v) {
    v4f r;
    r.x = fmaxf(v.x, 0.0f); r.y = fmaxf(v.y, 0.0f);
    r.z = fmaxf(v.z, 0.0f); r.w = fmaxf(v.w, 0.0f);
    return r;
}

// ---------------------------------------------------------------------------
// Tensor Data Mover: copy `ndwords` contiguous dwords from global -> LDS.
// D# layout per CDNA5 ISA (group0: count/lds/global/type,
// group1: data_size=4B, tensor_dim0=tile_dim0=ndwords, one row).
// 6-arg clang-23 builtin form: (u32x4, i32x8, i32x4, i32x4, i32x8, i32 cpol).
// ---------------------------------------------------------------------------
#if __has_builtin(__builtin_amdgcn_tensor_load_to_lds)
#define HAVE_TDM 1
typedef __attribute__((ext_vector_type(4))) uint32_t u32x4;
typedef __attribute__((ext_vector_type(8))) int      i32x8;
typedef __attribute__((ext_vector_type(4))) int      i32x4;

static __device__ __forceinline__ void
tdm_load_lds(const uint32_t* gsrc, uint32_t lds_off_bytes, uint32_t ndwords)
{
    const uint64_t ga = (uint64_t)(uintptr_t)gsrc;
    u32x4 g0;
    g0[0] = 1u;                                   // count=1 (valid), user mode
    g0[1] = lds_off_bytes;                        // lds_addr (bytes)
    g0[2] = (uint32_t)ga;                         // global_addr[31:0]
    g0[3] = (uint32_t)((ga >> 32) & 0x01FFFFFFu)  // global_addr[56:32]
          | (2u << 30);                           // type=2 ("image")
    i32x8 g1;
    g1[0] = (int)(2u << 16);                      // wg_mask=0, data_size=2 (4B)
    g1[1] = (int)((ndwords & 0xFFFFu) << 16);     // tensor_dim0[15:0]
    g1[2] = (int)(((ndwords >> 16) & 0xFFFFu)     // tensor_dim0[31:16]
          | (1u << 16));                          // tensor_dim1 = 1
    g1[3] = (int)((ndwords & 0xFFFFu) << 16);     // tile_dim0 = ndwords (<=65535)
    g1[4] = 1;                                    // tile_dim1=1, tile_dim2=0
    g1[5] = (int)ndwords;                         // tensor_dim0_stride lo
    g1[6] = 0;
    g1[7] = 0;
    const i32x4 z4 = {0, 0, 0, 0};                // groups 2/3 unused (2D)
    const i32x8 z8 = {0, 0, 0, 0, 0, 0, 0, 0};
    __builtin_amdgcn_tensor_load_to_lds(g0, g1, z4, z4, z8, 0);
}
#endif

// ---------------------------------------------------------------------------
// Pack A[M,K] (f32, row-major, optional ReLU) into per-lane WMMA fragments.
// 16-bit A 16x32 layout (ISA 7.12.2): lane = khalf*16 + row;
//   dword v: g=v>>2, K = kstep*32 + g*16 + khalf*8 + (v&3)*2 (pair packed).
// Storage: frag[((mtile*ksteps + kstep)*32 + lane)*8 + v]
// ---------------------------------------------------------------------------
__global__ void __launch_bounds__(256)
pack_a_frag(const float* __restrict__ A, uint32_t* __restrict__ out,
            int mtiles, int ksteps, int K, int relu)
{
    const int tid   = blockIdx.x * 256 + threadIdx.x;
    const int total = mtiles * ksteps * 32;
    if (tid >= total) return;
    const int lane  = tid & 31;
    const int tk    = tid >> 5;
    const int kstep = tk % ksteps;
    const int mtile = tk / ksteps;
    const int row   = lane & 15;
    const int khalf = lane >> 4;

    const float* __restrict__ src =
        A + (size_t)(mtile * 16 + row) * K + kstep * 32 + khalf * 8;
    v4f f0 = *(const v4f*)(src);
    v4f f1 = *(const v4f*)(src + 4);
    v4f f2 = *(const v4f*)(src + 16);
    v4f f3 = *(const v4f*)(src + 20);
    if (relu) { f0 = relu4(f0); f1 = relu4(f1); f2 = relu4(f2); f3 = relu4(f3); }

    v4u lo, hi;
    lo.x = pack2(f0.x, f0.y); lo.y = pack2(f0.z, f0.w);
    lo.z = pack2(f1.x, f1.y); lo.w = pack2(f1.z, f1.w);
    hi.x = pack2(f2.x, f2.y); hi.y = pack2(f2.z, f2.w);
    hi.z = pack2(f3.x, f3.y); hi.w = pack2(f3.z, f3.w);

    uint32_t* dst = out + ((size_t)(mtile * ksteps + kstep) * 32 + lane) * 8;
    *(v4u*)dst       = lo;
    *(v4u*)(dst + 4) = hi;
}

// Pack B[K,N] (f32) into fragments; N col = lane&15, K striping same as A.
__global__ void __launch_bounds__(256)
pack_b_frag(const float* __restrict__ B, uint32_t* __restrict__ out,
            int ntiles, int ksteps, int N)
{
    const int tid   = blockIdx.x * 256 + threadIdx.x;
    const int total = ntiles * ksteps * 32;
    if (tid >= total) return;
    const int lane  = tid & 31;
    const int tk    = tid >> 5;
    const int kstep = tk % ksteps;
    const int ntile = tk / ksteps;
    const int col   = ntile * 16 + (lane & 15);
    const int khalf = lane >> 4;
    const int k0    = kstep * 32;

    v4u lo, hi;
#pragma unroll
    for (int v = 0; v < 4; ++v) {
        const int ka = k0 + khalf * 8 + 2 * v;
        lo[v] = pack2(B[(size_t)ka * N + col], B[(size_t)(ka + 1) * N + col]);
        const int kb = k0 + 16 + khalf * 8 + 2 * v;
        hi[v] = pack2(B[(size_t)kb * N + col], B[(size_t)(kb + 1) * N + col]);
    }
    uint32_t* dst = out + ((size_t)(ntile * ksteps + kstep) * 32 + lane) * 8;
    *(v4u*)dst       = lo;
    *(v4u*)(dst + 4) = hi;
}

// ---------------------------------------------------------------------------
// GEMM: B fragments staged in LDS via TDM (one descriptor per block), then
// each wave computes a 16-row strip across all NT N-tiles. Inner loop:
// 2 global b128 (A) + 2*NT ds_load_b128 (B) + NT v_wmma per K-step, fully
// unrolled; DS latency is short so WMMAs no longer stall on global loads.
// ---------------------------------------------------------------------------
template <int NT, int KS>
__global__ void __launch_bounds__(256)
gemm_wmma_lds(const uint32_t* __restrict__ Ap, const uint32_t* __restrict__ Bp,
              const float* __restrict__ bias, float* __restrict__ out,
              int mtiles, int N)
{
    __shared__ __align__(16) uint32_t ldsB[NT * KS * 256];

    const int lane = threadIdx.x & 31;
    const int wid  = threadIdx.x >> 5;

#if HAVE_TDM
    if (wid == 0) {
        tdm_load_lds(Bp, 0u, (uint32_t)(NT * KS * 256));
        __builtin_amdgcn_s_wait_tensorcnt(0);
    }
#else
    for (int i = (int)threadIdx.x; i < NT * KS * 64; i += 256) {
        *(v4u*)(ldsB + (size_t)i * 4) = *(const v4u*)(Bp + (size_t)i * 4);
    }
#endif
    __syncthreads();

    const int mtile = blockIdx.x * 8 + wid;      // wave-uniform
    if (mtile >= mtiles) return;                 // uniform exit: EXEC all-1s

    const int col   = lane & 15;
    const int khalf = lane >> 4;

    v8f acc[NT];
#pragma unroll
    for (int nt = 0; nt < NT; ++nt) {
        const float bv = bias[nt * 16 + col];
#pragma unroll
        for (int r = 0; r < 8; ++r) acc[nt][r] = bv;
    }

    const uint32_t* __restrict__ ap = Ap + ((size_t)mtile * KS * 32 + lane) * 8;
    const uint32_t* bl = ldsB + lane * 8;

#pragma unroll
    for (int ks = 0; ks < KS; ++ks) {
        U8 at;
        at.lo = *(const v4u*)(ap + ks * 256);
        at.hi = *(const v4u*)(ap + ks * 256 + 4);
        const v16bf a = __builtin_bit_cast(v16bf, at);
#pragma unroll
        for (int nt = 0; nt < NT; ++nt) {
            const uint32_t* q = bl + (nt * KS + ks) * 256;
            U8 bt;
            bt.lo = *(const v4u*)q;
            bt.hi = *(const v4u*)(q + 4);
            const v16bf b = __builtin_bit_cast(v16bf, bt);
            // (neg_a, A, neg_b, B, c_mod, C, reuse_a, reuse_b)
            acc[nt] = __builtin_amdgcn_wmma_f32_16x16x32_bf16(
                false, a, false, b, (short)0, acc[nt], false, false);
        }
    }

    const int m0 = mtile * 16;
#pragma unroll
    for (int nt = 0; nt < NT; ++nt) {
#pragma unroll
        for (int r = 0; r < 8; ++r) {
            const int m = m0 + r + (khalf << 3);
            out[(size_t)m * N + nt * 16 + col] = acc[nt][r];
        }
    }
}

// ---------------------------------------------------------------------------
// Edge-parallel SpMM scatter: out[dst[e], :] += w[e] * Min[src[e], :].
// One thread per (edge, feature); F >= 32 pow2 => edge uniform per wave.
// fp32 atomic resolves in L2 (hardware global_atomic_add_f32).
// ---------------------------------------------------------------------------
__global__ void __launch_bounds__(256)
spmm_edge_scatter(const int* __restrict__ esrc, const int* __restrict__ edst,
                  const float* __restrict__ ew, const float* __restrict__ Min,
                  float* __restrict__ out, long long total, int logF)
{
    const long long tid = (long long)blockIdx.x * blockDim.x + threadIdx.x;
    if (tid >= total) return;
    const int e = (int)(tid >> logF);
    const int f = (int)(tid & ((1 << logF) - 1));
    const int s = esrc[e];
    const int d = edst[e];
    const float w = ew[e];
    const float v = w * Min[((size_t)s << logF) + f];
    unsafeAtomicAdd(&out[((size_t)d << logF) + f], v);
}

__global__ void __launch_bounds__(256)
zero_f32(float* __restrict__ p, long long n)
{
    const long long i = (long long)blockIdx.x * blockDim.x + threadIdx.x;
    if (i < n) p[i] = 0.0f;
}

// ---------------------------------------------------------------------------
// H1 = X@W1+b1 ; H = SpMM(H1) ; H2 = relu(H)@W2+b2 ; out = SpMM(H2)
// Workspace: R0 packed-A (51.2MB), R1 H1/H2 (51.2MB), R2 H (51.2MB),
//            R3 packed-B (80KB). Intermediates sized for the 192MB L2.
// ---------------------------------------------------------------------------
extern "C" void kernel_launch(void* const* d_in, const int* in_sizes, int n_in,
                              void* d_out, int out_size, void* d_ws, size_t ws_size,
                              hipStream_t stream)
{
    const float* X  = (const float*)d_in[0];
    const int*   es = (const int*)  d_in[1];
    const int*   ed = (const int*)  d_in[2];
    const float* ew = (const float*)d_in[3];
    const float* W1 = (const float*)d_in[4];
    const float* b1 = (const float*)d_in[5];
    const float* W2 = (const float*)d_in[6];
    const float* b2 = (const float*)d_in[7];

    const int IN_F = 256, HID = 128, OUT_F = 64;
    const int M = in_sizes[0] / IN_F;           // 100000 (multiple of 16)
    const long long E = in_sizes[1];            // 1600000
    const int mtiles = M / 16;                  // 6250
    const int ks1 = IN_F / 32, ks2 = HID / 32;  // 8, 4
    const int nt1 = HID / 16,  nt2 = OUT_F / 16;// 8, 4

    uint8_t* ws = (uint8_t*)d_ws;
    const size_t szR = (size_t)mtiles * ks1 * 32 * 8 * 4;   // 51.2 MB
    uint32_t* Apack = (uint32_t*)(ws);                      // R0
    float*    H1    = (float*)   (ws + szR);                // R1
    float*    H2    = H1;                                   // R1 (H1 dead)
    float*    H     = (float*)   (ws + 2 * szR);            // R2
    uint32_t* Bp1   = (uint32_t*)(ws + 3 * szR);            // R3
    uint32_t* Bp2   = Bp1 + (size_t)nt1 * ks1 * 32 * 8;

    // --- Stage 1: H1 = X @ W1 + b1 ---
    {
        const int ta = mtiles * ks1 * 32;
        pack_a_frag<<<(ta + 255) / 256, 256, 0, stream>>>(X, Apack, mtiles, ks1, IN_F, 0);
        const int tb = nt1 * ks1 * 32;
        pack_b_frag<<<(tb + 255) / 256, 256, 0, stream>>>(W1, Bp1, nt1, ks1, HID);
        gemm_wmma_lds<8, 8><<<(mtiles + 7) / 8, 256, 0, stream>>>(
            Apack, Bp1, b1, H1, mtiles, HID);
    }
    // --- Stage 2: H = scatter-add(w * H1[src]) ---
    {
        const long long n = (long long)M * HID;
        zero_f32<<<(unsigned)((n + 255) / 256), 256, 0, stream>>>(H, n);
        const long long total = E << 7;
        spmm_edge_scatter<<<(unsigned)((total + 255) / 256), 256, 0, stream>>>(
            es, ed, ew, H1, H, total, 7);
    }
    // --- Stage 3: H2 = relu(H) @ W2 + b2 (ReLU fused into packing) ---
    {
        const int ta = mtiles * ks2 * 32;
        pack_a_frag<<<(ta + 255) / 256, 256, 0, stream>>>(H, Apack, mtiles, ks2, HID, 1);
        const int tb = nt2 * ks2 * 32;
        pack_b_frag<<<(tb + 255) / 256, 256, 0, stream>>>(W2, Bp2, nt2, ks2, OUT_F);
        gemm_wmma_lds<4, 4><<<(mtiles + 7) / 8, 256, 0, stream>>>(
            Apack, Bp2, b2, H2, mtiles, OUT_F);
    }
    // --- Stage 4: out = scatter-add(w * H2[src]) ---
    {
        float* outp = (float*)d_out;
        const long long n = (long long)M * OUT_F;
        zero_f32<<<(unsigned)((n + 255) / 256), 256, 0, stream>>>(outp, n);
        const long long total = E << 6;
        spmm_edge_scatter<<<(unsigned)((total + 255) / 256), 256, 0, stream>>>(
            es, ed, ew, H2, outp, total, 6);
    }
}
